// TransformerMultiHeadAttention_76931454206233
// MI455X (gfx1250) — matrix operand order repaired and checked by reading, not compiled, not run on hardware
//
#include <hip/hip_runtime.h>
#include <hip/hip_bf16.h>

// ---------------------------------------------------------------------------
// Problem constants
// ---------------------------------------------------------------------------
#define B_      32
#define T_      512
#define C_      768
#define H_      12
#define D_      64
#define THREE_D 192
#define BT      (B_ * T_)   // 16384
#define HC      (H_ * C_)   // 9216

typedef __attribute__((ext_vector_type(16))) __bf16 v16bf;
typedef __attribute__((ext_vector_type(8)))  float  v8f;

union FragBF {
    v16bf f;
    uint4 u[2];
};

// ---------------------------------------------------------------------------
// CDNA5 async global->LDS copy (ASYNCcnt-tracked), inline asm per ISA
// GLOBAL_LOAD_ASYNC_TO_LDS_B128: vdst = LDS byte address, vaddr = 64-bit addr.
// LDS offset = low 32 bits of the flat shared-memory address (aperture map).
// ---------------------------------------------------------------------------
__device__ __forceinline__ void async_copy_b128(const void* gptr, void* lds_ptr)
{
    unsigned loff = (unsigned)(unsigned long long)lds_ptr;
    asm volatile("global_load_async_to_lds_b128 %0, %1, off"
                 :: "v"(loff), "v"(gptr) : "memory");
}

__device__ __forceinline__ void wait_async0()
{
    asm volatile("s_wait_asynccnt 0" ::: "memory");
}

// ---------------------------------------------------------------------------
// Shared WMMA GEMM tile:  C[M,N] = alpha * A[M,K] * B[K,N] + bias
// A row-major (lda); B supplied TRANSPOSED as Bt[N,K] row-major (ldbt) so all
// WMMA fragments are contiguous 16-byte LDS loads.
// Workgroup tile: 256(M) x 64(N), K-step 32, 256 threads = 8 waves (wave32).
// Wave w owns 2 M-subtiles (rows [32w,32w+32)) x 4 N-subtiles -> 8 acc tiles.
// LDS double-buffered; tiles staged with async global->LDS b128 copies.
// ---------------------------------------------------------------------------
#define BLK_M   256
#define BLK_N   64
#define BLK_K   32
#define MSUB    2
#define NSUB    (BLK_N / 16)
#define THREADS 256
#define LDS_PAD 8   // row stride 40 bf16 = 80 B, keeps 16B alignment

template <bool OUT_BF16>
__device__ __forceinline__ void gemm_tile(
    const __bf16* __restrict__ A,  int lda,
    const __bf16* __restrict__ Bt, int ldbt,
    void* __restrict__ Cout,       int ldc,
    const float* __restrict__ bias,
    float alpha, int K)
{
    __shared__ __bf16 As[2][BLK_M][BLK_K + LDS_PAD];
    __shared__ __bf16 Bs[2][BLK_N][BLK_K + LDS_PAD];

    const int tid  = threadIdx.x;
    const int lane = tid & 31;
    const int wave = tid >> 5;     // 0..7
    const int r16  = lane & 15;    // row/col within 16
    const int hk   = lane >> 4;    // lane half selects K sub-range

    v8f acc[MSUB][NSUB];
#pragma unroll
    for (int ms = 0; ms < MSUB; ++ms)
#pragma unroll
        for (int j = 0; j < NSUB; ++j)
            acc[ms][j] = (v8f){0.f, 0.f, 0.f, 0.f, 0.f, 0.f, 0.f, 0.f};

    // Per-thread staging map: thread t copies A row t (4 x 16B) and one B chunk
    const int brow = tid >> 2;
    const int bkk  = (tid & 3) * 8;

    // Prologue: stage tile 0 into buffer 0
#pragma unroll
    for (int i = 0; i < 4; ++i)
        async_copy_b128(&A[(size_t)tid * lda + i * 8], &As[0][tid][i * 8]);
    async_copy_b128(&Bt[(size_t)brow * ldbt + bkk], &Bs[0][brow][bkk]);

    const int nsteps = K / BLK_K;
    for (int s = 0; s < nsteps; ++s) {
        const int buf = s & 1;

        wait_async0();          // this wave's copies for tile s have landed
        __syncthreads();        // ... and so have every other wave's

        // Stage tile s+1 into the other buffer (overlaps with compute below).
        // Safe: all waves passed the barrier, so reads of buf^1 (tile s-1)
        // are complete before these copies can overwrite it.
        if (s + 1 < nsteps) {
            const int k1 = (s + 1) * BLK_K;
#pragma unroll
            for (int i = 0; i < 4; ++i)
                async_copy_b128(&A[(size_t)tid * lda + k1 + i * 8],
                                &As[buf ^ 1][tid][i * 8]);
            async_copy_b128(&Bt[(size_t)brow * ldbt + k1 + bkk],
                            &Bs[buf ^ 1][brow][bkk]);
        }

        // Compute from buffer `buf`.
#pragma unroll
        for (int ms = 0; ms < MSUB; ++ms) {
            const int arow = (wave * MSUB + ms) * 16 + r16;
            // A fragment (16x32): lanes 0-15 hold K 0-7 & 16-23,
            // lanes 16-31 hold K 8-15 & 24-31.
            FragBF a;
            a.u[0] = *(const uint4*)&As[buf][arow][hk * 8];
            a.u[1] = *(const uint4*)&As[buf][arow][16 + hk * 8];
#pragma unroll
            for (int j = 0; j < NSUB; ++j) {
                // B fragment (32x16): lane holds column n = lane&15;
                // lanes 0-15: K 0-7 & 16-23, lanes 16-31: K 8-15 & 24-31.
                FragBF b;
                b.u[0] = *(const uint4*)&Bs[buf][j * 16 + r16][hk * 8];
                b.u[1] = *(const uint4*)&Bs[buf][j * 16 + r16][16 + hk * 8];

                acc[ms][j] = __builtin_amdgcn_wmma_f32_16x16x32_bf16(
                    /*neg_a=*/false, a.f, /*neg_b=*/false, b.f,
                    /*c_mod=*/(short)0, acc[ms][j],
                    /*reuse_a=*/false, /*reuse_b=*/false);
            }
        }
        __syncthreads();  // tile s fully consumed before next overwrite cycle
    }

    // Epilogue: C/D layout = 8 VGPRs, row = 8*(lane>=16) + vgpr, col = lane&15
#pragma unroll
    for (int ms = 0; ms < MSUB; ++ms) {
#pragma unroll
        for (int j = 0; j < NSUB; ++j) {
            int   col  = j * 16 + r16;
            float bval = bias ? bias[col] : 0.f;
#pragma unroll
            for (int i = 0; i < 8; ++i) {
                int   row = (wave * MSUB + ms) * 16 + hk * 8 + i;
                float v   = acc[ms][j][i] * alpha + bval;
                if (OUT_BF16)
                    ((__bf16*)Cout)[(size_t)row * ldc + col] = (__bf16)v;
                else
                    ((float*)Cout)[(size_t)row * ldc + col] = v;
            }
        }
    }
}

// ---------------------------------------------------------------------------
// Conversion / layout kernels
// ---------------------------------------------------------------------------
__global__ __launch_bounds__(256) void k_convert_f32_bf16(
    const float* __restrict__ in, __bf16* __restrict__ out, size_t n)
{
    size_t i = (size_t)blockIdx.x * 256 + threadIdx.x;
    if (i < n) out[i] = (__bf16)in[i];
}

// in:  f32 [batch][R][Cc]  ->  out: bf16 [batch][Cc][R]   (B-operand pre-transpose)
__global__ __launch_bounds__(256) void k_transpose_convert(
    const float* __restrict__ in, __bf16* __restrict__ out, int R, int Cc)
{
    size_t n = (size_t)R * Cc;
    size_t i = (size_t)blockIdx.x * 256 + threadIdx.x;
    if (i >= n) return;
    int c = (int)(i % Cc);
    int r = (int)(i / Cc);
    const float* src = in + (size_t)blockIdx.z * n;
    __bf16* dst      = out + (size_t)blockIdx.z * n;
    dst[(size_t)c * R + r] = (__bf16)src[i];
}

// Build Vt[z][d][t] from qkv[h][(b*T+t)*192 + 128 + d], z = b*H + h
__global__ __launch_bounds__(256) void k_build_vT(
    const __bf16* __restrict__ qkv, __bf16* __restrict__ vT)
{
    size_t i  = (size_t)blockIdx.x * 256 + threadIdx.x;  // over 384*64*512
    int    t  = (int)(i & (T_ - 1));
    size_t zd = i >> 9;
    int    d  = (int)(zd & (D_ - 1));
    size_t z  = zd >> 6;
    int    b  = (int)(z / H_);
    int    h  = (int)(z % H_);
    vT[i] = qkv[(size_t)h * BT * THREE_D + ((size_t)b * T_ + t) * THREE_D + 128 + d];
}

// Row softmax over T_=512 cols; normalizes f32 in place, writes bf16 copy.
__global__ __launch_bounds__(256) void k_softmax(
    float* __restrict__ probs, __bf16* __restrict__ pbf)
{
    size_t  row = blockIdx.x;                // B*H*T rows
    float*  p   = probs + row * T_;
    __bf16* q   = pbf + row * T_;
    int     t   = threadIdx.x;

    float v0 = p[t], v1 = p[t + 256];
    __shared__ float red[256];

    red[t] = fmaxf(v0, v1);
    __syncthreads();
    for (int s = 128; s > 0; s >>= 1) {
        if (t < s) red[t] = fmaxf(red[t], red[t + s]);
        __syncthreads();
    }
    float m = red[0];
    __syncthreads();

    float e0 = __expf(v0 - m), e1 = __expf(v1 - m);
    red[t] = e0 + e1;
    __syncthreads();
    for (int s = 128; s > 0; s >>= 1) {
        if (t < s) red[t] += red[t + s];
        __syncthreads();
    }
    float inv = 1.0f / red[0];

    float o0 = e0 * inv, o1 = e1 * inv;
    p[t] = o0;         p[t + 256] = o1;
    q[t] = (__bf16)o0; q[t + 256] = (__bf16)o1;
}

// ---------------------------------------------------------------------------
// GEMM wrappers
// ---------------------------------------------------------------------------
__global__ __launch_bounds__(256) void k_gemm_qkv(
    const __bf16* __restrict__ x, const __bf16* __restrict__ WqkvT,
    const float* __restrict__ bqkv, __bf16* __restrict__ qkv)
{
    int n0 = blockIdx.x * BLK_N;
    int m0 = blockIdx.y * BLK_M;
    int h  = blockIdx.z;
    gemm_tile<true>(x + (size_t)m0 * C_, C_,
                    WqkvT + (size_t)h * THREE_D * C_ + (size_t)n0 * C_, C_,
                    qkv + (size_t)h * BT * THREE_D + (size_t)m0 * THREE_D + n0, THREE_D,
                    bqkv + h * THREE_D + n0, 1.0f, C_);
}

__global__ __launch_bounds__(256) void k_gemm_scores(
    const __bf16* __restrict__ qkv, float* __restrict__ probs)
{
    int n0 = blockIdx.x * BLK_N;
    int m0 = blockIdx.y * BLK_M;
    int z  = blockIdx.z;
    int b  = z / H_, h = z % H_;
    const __bf16* base = qkv + (size_t)h * BT * THREE_D + (size_t)b * T_ * THREE_D;
    gemm_tile<false>(base + (size_t)m0 * THREE_D, THREE_D,       // Q rows
                     base + 64 + (size_t)n0 * THREE_D, THREE_D,  // K rows (=Bt)
                     probs + (size_t)z * T_ * T_ + (size_t)m0 * T_ + n0, T_,
                     nullptr, 0.125f /* 1/sqrt(64) */, D_);
}

__global__ __launch_bounds__(256) void k_gemm_pv(
    const __bf16* __restrict__ pbf, const __bf16* __restrict__ vT,
    __bf16* __restrict__ sa)
{
    int m0 = blockIdx.y * BLK_M;      // N = 64 -> single n-block
    int z  = blockIdx.z;
    gemm_tile<true>(pbf + (size_t)z * T_ * T_ + (size_t)m0 * T_, T_,
                    vT + (size_t)z * D_ * T_, T_,
                    sa + (size_t)z * T_ * D_ + (size_t)m0 * D_, D_,
                    nullptr, 1.0f, T_);
}

__global__ __launch_bounds__(256) void k_gemm_headproj(
    const __bf16* __restrict__ sa, const __bf16* __restrict__ WphT,
    const float* __restrict__ bph, __bf16* __restrict__ cat)
{
    int n0 = blockIdx.x * BLK_N;
    int m0 = blockIdx.y * BLK_M;
    int z  = blockIdx.z;
    int b  = z / H_, h = z % H_;
    gemm_tile<true>(sa + (size_t)z * T_ * D_ + (size_t)m0 * D_, D_,
                    WphT + (size_t)h * C_ * D_ + (size_t)n0 * D_, D_,
                    cat + ((size_t)b * T_ + m0) * HC + (size_t)h * C_ + n0, HC,
                    bph + h * C_ + n0, 1.0f, D_);
}

__global__ __launch_bounds__(256) void k_gemm_final(
    const __bf16* __restrict__ cat, const __bf16* __restrict__ WprojT,
    const float* __restrict__ bproj, float* __restrict__ out)
{
    int n0 = blockIdx.x * BLK_N;
    int m0 = blockIdx.y * BLK_M;
    gemm_tile<false>(cat + (size_t)m0 * HC, HC,
                     WprojT + (size_t)n0 * HC, HC,
                     out + (size_t)m0 * C_ + n0, C_,
                     bproj + n0, 1.0f, HC);
}

// ---------------------------------------------------------------------------
// Host-side launch
// ---------------------------------------------------------------------------
extern "C" void kernel_launch(void* const* d_in, const int* in_sizes, int n_in,
                              void* d_out, int out_size, void* d_ws, size_t ws_size,
                              hipStream_t stream)
{
    const float* x     = (const float*)d_in[0];
    const float* Wqkv  = (const float*)d_in[1];
    const float* bqkv  = (const float*)d_in[2];
    const float* Wph   = (const float*)d_in[3];
    const float* bph   = (const float*)d_in[4];
    const float* Wproj = (const float*)d_in[5];
    const float* bproj = (const float*)d_in[6];

    float* sa_out = (float*)d_out;                       // [BT, C]
    float* probs  = sa_out + (size_t)BT * C_;            // [B*H, T, T] f32

    // Workspace layout (bf16 elements)
    __bf16* ws = (__bf16*)d_ws;
    size_t  o  = 0;
    __bf16* x_bf    = ws + o; o += (size_t)BT * C_;          // 16384*768
    __bf16* WqkvT   = ws + o; o += (size_t)H_ * THREE_D * C_;
    __bf16* WphT    = ws + o; o += (size_t)H_ * C_ * D_;
    __bf16* WprojT  = ws + o; o += (size_t)C_ * HC;
    __bf16* qkv     = ws + o; o += (size_t)H_ * BT * THREE_D;
    __bf16* vT      = ws + o; o += (size_t)B_ * H_ * D_ * T_;
    __bf16* pbf     = ws + o; o += (size_t)B_ * H_ * T_ * T_;
    __bf16* sa      = ws + o; o += (size_t)B_ * H_ * T_ * D_;
    __bf16* cat     = ws + o; o += (size_t)BT * HC;

    // --- 0. conversions / pre-transposes -----------------------------------
    {
        size_t n = (size_t)BT * C_;
        k_convert_f32_bf16<<<dim3((unsigned)((n + 255) / 256)), 256, 0, stream>>>(x, x_bf, n);
    }
    k_transpose_convert<<<dim3((C_ * THREE_D + 255) / 256, 1, H_), 256, 0, stream>>>(
        Wqkv, WqkvT, C_, THREE_D);
    k_transpose_convert<<<dim3((D_ * C_ + 255) / 256, 1, H_), 256, 0, stream>>>(
        Wph, WphT, D_, C_);
    k_transpose_convert<<<dim3((HC * C_ + 255) / 256, 1, 1), 256, 0, stream>>>(
        Wproj, WprojT, HC, C_);

    // --- 1. QKV projection: per-head [16384,768] x [768,192] ---------------
    k_gemm_qkv<<<dim3(THREE_D / BLK_N, BT / BLK_M, H_), THREADS, 0, stream>>>(
        x_bf, WqkvT, bqkv, qkv);

    // --- 2. V transpose for PV GEMM ----------------------------------------
    {
        size_t n = (size_t)B_ * H_ * D_ * T_;
        k_build_vT<<<dim3((unsigned)(n / 256)), 256, 0, stream>>>(qkv, vT);
    }

    // --- 3. Scores: per (b,h) [512,64] x [64,512] * 1/sqrt(D) --------------
    k_gemm_scores<<<dim3(T_ / BLK_N, T_ / BLK_M, B_ * H_), THREADS, 0, stream>>>(
        qkv, probs);

    // --- 4. Softmax (f32 in place into d_out, bf16 copy for PV) ------------
    k_softmax<<<dim3(B_ * H_ * T_), 256, 0, stream>>>(probs, pbf);

    // --- 5. PV: per (b,h) [512,512] x [512,64] -----------------------------
    k_gemm_pv<<<dim3(1, T_ / BLK_M, B_ * H_), THREADS, 0, stream>>>(pbf, vT, sa);

    // --- 6. Per-head out-proj into concatenated layout ---------------------
    k_gemm_headproj<<<dim3(C_ / BLK_N, T_ / BLK_M, B_ * H_), THREADS, 0, stream>>>(
        sa, WphT, bph, cat);

    // --- 7. Final projection: [16384,9216] x [9216,768] --------------------
    k_gemm_final<<<dim3(C_ / BLK_N, BT / BLK_M, 1), THREADS, 0, stream>>>(
        cat, WprojT, bproj, sa_out);
}
